// ViT_83227876262151
// MI455X (gfx1250) — compile-verified, matrix operand order checked by reading, not run on hardware
//
#include <hip/hip_runtime.h>
#include <hip/hip_bf16.h>
#include <math.h>
#include <stdint.h>

// ---------------------------------------------------------------------------
// ViT forward for MI455X (gfx1250, wave32, WMMA 16x16x32 f16 -> f32 acc,
// Tensor Data Mover staging of weight tiles into LDS).
// ---------------------------------------------------------------------------

typedef __attribute__((ext_vector_type(16))) _Float16 v16h;
typedef __attribute__((ext_vector_type(8)))  _Float16 v8h;
typedef __attribute__((ext_vector_type(8)))  float    v8f;
typedef __attribute__((ext_vector_type(4)))  unsigned int u32x4;
typedef __attribute__((ext_vector_type(8)))  int          i32x8;
typedef __attribute__((ext_vector_type(4)))  int          i32x4;

namespace vit {
constexpr int B     = 256;
constexpr int C     = 3;
constexpr int IMG   = 224;
constexpr int P     = 16;
constexpr int G     = IMG / P;      // 14
constexpr int NP    = G * G;        // 196
constexpr int PD    = C * P * P;    // 768
constexpr int E     = 256;
constexpr int H     = 4;
constexpr int D     = 64;
constexpr int DEPTH = 8;
constexpr int MLP   = 256;
constexpr int SEQ   = NP + 1;       // 197
constexpr int NCLS  = 10;
constexpr int MPATCH = B * NP;      // 50176 rows (mult of 128)
constexpr int MTOK   = B * SEQ;     // 50432 rows (mult of 128)
}

#if defined(__HIP_DEVICE_COMPILE__) && defined(__gfx1250__) && \
    __has_builtin(__builtin_amdgcn_tensor_load_to_lds) && \
    __has_builtin(__builtin_amdgcn_s_wait_tensorcnt)
#define VIT_USE_TDM 1
#else
#define VIT_USE_TDM 0
#endif

// ---------------------------------------------------------------------------
// WMMA fragment helpers (CDNA5 16-bit A/B layout, wave32).
// Element h of the v16h in lane L maps to:
//   row = L % 16
//   k   = kbase + 8*(L/16) + (h<8 ? h : 16 + (h-8))
// => two contiguous 8-half (16B) runs per lane when K is the contiguous axis.
// ---------------------------------------------------------------------------
union Frag16 { v16h v; v8h h[2]; };

__device__ __forceinline__ v16h frag_load(const _Float16* __restrict__ base,
                                          int ld, int row0, int kbase,
                                          int lane, int rowmax) {
  const int row = row0 + (lane & 15);
  const int k0  = kbase + ((lane >> 4) << 3);
  Frag16 f;
  if (row < rowmax) {
    const _Float16* p = base + (size_t)row * ld + k0;
    f.h[0] = *(const v8h*)(p);
    f.h[1] = *(const v8h*)(p + 16);
  } else {
    v8h z;
    #pragma unroll
    for (int i = 0; i < 8; ++i) z[i] = (_Float16)0.0f;
    f.h[0] = z; f.h[1] = z;
  }
  return f.v;
}

__device__ __forceinline__ v8f wmma_f16(v16h a, v16h b, v8f c) {
  return __builtin_amdgcn_wmma_f32_16x16x32_f16(
      /*neg_a=*/false, a, /*neg_b=*/false, b,
      /*c_mod=*/(short)0, c, /*reuse_a=*/false, /*reuse_b=*/false);
}

// ---------------------------------------------------------------------------
// TDM: DMA a [rows_valid(<=64)] x [32 half] tile of W (row stride ldw elems)
// into LDS at lds_byte_addr, zero-filling rows >= rows_valid (TDM OOB reads
// return zero).  D# layout per CDNA5 ISA ch.8 (2D tensor, groups 2/3 zero).
// ---------------------------------------------------------------------------
#if VIT_USE_TDM
__device__ __forceinline__ void tdm_load_b_tile(const _Float16* gsrc, int ldw,
                                                int rows_valid,
                                                unsigned lds_byte_addr) {
  const unsigned long long ga = (unsigned long long)(uintptr_t)gsrc;
  u32x4 g0;
  g0[0] = 1u;                                  // count=1 (valid), user mode
  g0[1] = lds_byte_addr;                       // lds_addr
  g0[2] = (unsigned)(ga & 0xffffffffu);        // global_addr[31:0]
  g0[3] = (unsigned)((ga >> 32) & 0x01ffffffu) // global_addr[56:32]
          | 0x80000000u;                       // type=2 ("image") in [127:126]
  i32x8 g1;
  g1[0] = (int)(1u << 16);                     // data_size=1 (2 bytes), mask=0
  g1[1] = (int)(32u << 16);                    // tensor_dim0=32 (bits 79:48)
  g1[2] = (int)(((unsigned)rows_valid & 0xffffu) << 16);  // tensor_dim1 lo16
  g1[3] = (int)(32u << 16);                    // tile_dim0=32 (bits 127:112)
  g1[4] = 64;                                  // tile_dim1=64, tile_dim2=0
  g1[5] = ldw;                                 // tensor_dim0_stride[31:0]
  g1[6] = 0;                                   // stride hi / dim1_stride lo
  g1[7] = 0;
  i32x4 gz;
  gz[0] = 0; gz[1] = 0; gz[2] = 0; gz[3] = 0;
#if __has_include(<hip/amd_detail/amd_gfx1250_TDM.h>)
  i32x8 gz8;
  #pragma unroll
  for (int i = 0; i < 8; ++i) gz8[i] = 0;
  __builtin_amdgcn_tensor_load_to_lds(g0, g1, gz, gz, gz8, 0);
#else
  __builtin_amdgcn_tensor_load_to_lds(g0, g1, gz, gz, 0);
#endif
  __builtin_amdgcn_s_wait_tensorcnt(0);
}
#endif

// ---------------------------------------------------------------------------
// Generic GEMM: Cout[M,N] = A[M,K] @ W[N,K]^T + bias.
// Block tile 128(M) x 64(N): 8 waves, each wave owns 16 rows and all 4
// 16-wide N tiles (4 accumulators -> 4 WMMAs per staged k-chunk).
// Weight chunk [64 x 32] staged in LDS per k-step (TDM if available).
// M must be a multiple of 128; N may be ragged (guarded by wrows / N).
// ---------------------------------------------------------------------------
template <bool OUT_F16>
__global__ void __launch_bounds__(256)
k_gemm(const _Float16* __restrict__ A, int lda,
       const _Float16* __restrict__ W, int ldw, int wrows,
       const float* __restrict__ bias,
       void* __restrict__ Cout, int ldc, int K, int N) {
  __shared__ _Float16 Bs[64 * 32];             // 4 KB: one staged k-chunk
  const int tid  = threadIdx.x;
  const int lane = tid & 31;
  const int wave = tid >> 5;
  const int m0  = (blockIdx.x * 8 + wave) * 16;
  const int bn0 = blockIdx.y * 64;
  const int rows_valid = (wrows - bn0 < 64) ? (wrows - bn0) : 64;

  v8f acc[4];
  #pragma unroll
  for (int j = 0; j < 4; ++j)
    #pragma unroll
    for (int r = 0; r < 8; ++r) acc[j][r] = 0.0f;

  for (int k = 0; k < K; k += 32) {
    __syncthreads();
#if VIT_USE_TDM
    if (tid < 32) {   // wave 0 issues the DMA; descriptor is wave-uniform
      tdm_load_b_tile(W + (size_t)bn0 * ldw + k, ldw, rows_valid,
                      (unsigned)(size_t)(void*)Bs);
    }
#else
    {
      // cooperative fallback: 256 threads x 16B = 4 KB
      const int row = tid >> 2;
      const int kc  = (tid & 3) * 8;
      v8h val;
      if (row < rows_valid) {
        val = *(const v8h*)(W + (size_t)(bn0 + row) * ldw + k + kc);
      } else {
        #pragma unroll
        for (int i = 0; i < 8; ++i) val[i] = (_Float16)0.0f;
      }
      *(v8h*)&Bs[row * 32 + kc] = val;
    }
#endif
    __syncthreads();

    // speculative prefetch of the next A chunk (global_prefetch_b8)
    __builtin_prefetch((const void*)(A + (size_t)(m0 + (lane & 15)) * lda + k + 32), 0, 0);
    v16h a = frag_load(A, lda, m0, k, lane, 0x7fffffff);
    #pragma unroll
    for (int j = 0; j < 4; ++j) {
      v16h b = frag_load(Bs, 32, j * 16, 0, lane, 1 << 30);
      acc[j] = wmma_f16(a, b, acc[j]);
    }
  }

  const int rbase = m0 + ((lane >> 4) << 3);
  #pragma unroll
  for (int j = 0; j < 4; ++j) {
    const int col = bn0 + j * 16 + (lane & 15);
    if (col < N) {
      const float bv = bias ? bias[col] : 0.0f;
      #pragma unroll
      for (int r = 0; r < 8; ++r) {
        const float v = acc[j][r] + bv;
        const size_t idx = (size_t)(rbase + r) * ldc + col;
        if (OUT_F16) ((_Float16*)Cout)[idx] = (_Float16)v;
        else         ((float*)Cout)[idx]   = v;
      }
    }
  }
}

// ---------------------------------------------------------------------------
// im2col: x[B,C,224,224] -> patches f16 [B*196, 768] (K-contiguous)
// ---------------------------------------------------------------------------
__global__ void __launch_bounds__(256)
k_im2col(const float* __restrict__ x, _Float16* __restrict__ p16) {
  using namespace vit;
  const int idx = blockIdx.x * 256 + threadIdx.x;   // exact: B*NP*PD / 256
  const int col = idx % PD;
  const int m   = idx / PD;
  const int b = m / NP, p = m % NP;
  const int c = col >> 8, r = col & 255;
  const int py = r >> 4, px = r & 15;
  const int gy = p / G,  gx = p % G;
  const size_t src = (((size_t)b * C + c) * IMG + gy * P + py) * IMG + gx * P + px;
  p16[idx] = (_Float16)x[src];
}

// f32 -> f16 conversion (weights etc.)
__global__ void __launch_bounds__(256)
k_cvt(const float* __restrict__ src, _Float16* __restrict__ dst, int n) {
  const int i = blockIdx.x * 256 + threadIdx.x;
  if (i < n) dst[i] = (_Float16)src[i];
}

// emb = concat(cls, patch_embed) + pos ; also keeps an f16 mirror
__global__ void __launch_bounds__(256)
k_embed(const float* __restrict__ pe, const float* __restrict__ cls,
        const float* __restrict__ pos, float* __restrict__ emb,
        _Float16* __restrict__ emb16) {
  using namespace vit;
  const int idx = blockIdx.x * 256 + threadIdx.x;   // exact: MTOK*E / 256
  const int e = idx & 255;
  const int m = idx >> 8;
  const int b = m / SEQ, s = m % SEQ;
  float v;
  if (s == 0) v = cls[e] + pos[e];
  else        v = pe[(size_t)(b * NP + s - 1) * E + e] + pos[s * E + e];
  emb[idx] = v;
  emb16[idx] = (_Float16)v;
}

// ---------------------------------------------------------------------------
// Fused attention for one (n, h, 64-query-block).
// Reference semantics: attention runs over the IMAGE-BATCH axis (q,k in
// [0,256)), with (n,h) as batch dims.  Per block:
//   Q[64,64], K[256,64], V^T[64,256] staged in LDS (f16),
//   S = QK^T * 0.125 (WMMA, f32 in LDS), row softmax -> f16 P,
//   O = P V (WMMA) -> o16[b, n, h*64+d].
// Dynamic LDS: 8K Qs + 32K Ks + 32K Vt + 32K S16 + 64K Sf = 168 KB.
// ---------------------------------------------------------------------------
__global__ void __launch_bounds__(256)
k_attn(const _Float16* __restrict__ qkv, _Float16* __restrict__ o16) {
  using namespace vit;
  extern __shared__ char smem[];
  _Float16* Qs  = (_Float16*)(smem);            // [64][64]
  _Float16* Ks  = (_Float16*)(smem + 8192);     // [256][64]
  _Float16* Vt  = (_Float16*)(smem + 40960);    // [64][256]  (d-major)
  _Float16* S16 = (_Float16*)(smem + 73728);    // [64][256]
  float*    Sf  = (float*)   (smem + 106496);   // [64][256]

  const int nh = blockIdx.x;
  const int n  = nh >> 2;
  const int h  = nh & 3;
  const int qb = blockIdx.y;                    // 64-row query block
  const int tid  = threadIdx.x;
  const int lane = tid & 31;
  const int wave = tid >> 5;

  // --- stage K, V^T, Q ---
  for (int i = tid; i < 256 * 64; i += 256) {
    const int kk = i >> 6, d = i & 63;
    Ks[i] = qkv[((size_t)kk * SEQ + n) * (3 * E) + E + h * 64 + d];
  }
  for (int i = tid; i < 64 * 256; i += 256) {
    const int d = i >> 8, kk = i & 255;
    Vt[i] = qkv[((size_t)kk * SEQ + n) * (3 * E) + 2 * E + h * 64 + d];
  }
  for (int i = tid; i < 64 * 64; i += 256) {
    const int r = i >> 6, d = i & 63;
    Qs[i] = qkv[((size_t)(qb * 64 + r) * SEQ + n) * (3 * E) + h * 64 + d];
  }
  __syncthreads();

  // --- S = Q K^T * scale : 4x16 grid of 16x16 tiles, 8 per wave ---
  for (int t = wave; t < 64; t += 8) {
    const int tm = t >> 4, tn = t & 15;
    v8f c;
    #pragma unroll
    for (int r = 0; r < 8; ++r) c[r] = 0.0f;
    #pragma unroll
    for (int k = 0; k < 64; k += 32) {
      v16h a = frag_load(Qs, 64, tm * 16, k, lane, 1 << 30);
      v16h b = frag_load(Ks, 64, tn * 16, k, lane, 1 << 30);
      c = wmma_f16(a, b, c);
    }
    const int col   = tn * 16 + (lane & 15);
    const int rbase = tm * 16 + ((lane >> 4) << 3);
    #pragma unroll
    for (int r = 0; r < 8; ++r) Sf[(rbase + r) * 256 + col] = c[r] * 0.125f;
  }
  __syncthreads();

  // --- row softmax (64 rows, one thread each) -> f16 probabilities ---
  if (tid < 64) {
    const float* srow = Sf + tid * 256;
    float m = -1e30f;
    for (int j = 0; j < 256; ++j) m = fmaxf(m, srow[j]);
    float s = 0.0f;
    for (int j = 0; j < 256; ++j) s += expf(srow[j] - m);
    const float inv = 1.0f / s;
    _Float16* drow = S16 + tid * 256;
    for (int j = 0; j < 256; ++j) drow[j] = (_Float16)(expf(srow[j] - m) * inv);
  }
  __syncthreads();

  // --- O = P V : 4x4 grid of 16x16 tiles, 2 per wave ---
  for (int t = wave; t < 16; t += 8) {
    const int tm = t >> 2, tn = t & 3;
    v8f c;
    #pragma unroll
    for (int r = 0; r < 8; ++r) c[r] = 0.0f;
    #pragma unroll
    for (int k = 0; k < 256; k += 32) {
      v16h a = frag_load(S16, 256, tm * 16, k, lane, 1 << 30);
      v16h b = frag_load(Vt,  256, tn * 16, k, lane, 1 << 30);
      c = wmma_f16(a, b, c);
    }
    const int col   = tn * 16 + (lane & 15);          // d within head
    const int rbase = tm * 16 + ((lane >> 4) << 3);
    #pragma unroll
    for (int r = 0; r < 8; ++r) {
      const int brow = qb * 64 + rbase + r;           // image-batch index
      o16[((size_t)brow * SEQ + n) * E + h * 64 + col] = (_Float16)c[r];
    }
  }
}

// emb = LayerNorm(emb + add) * g + b ; also writes f16 mirror. Block per row.
__global__ void __launch_bounds__(256)
k_ln_residual(float* __restrict__ emb, const float* __restrict__ add,
              const float* __restrict__ g, const float* __restrict__ b,
              _Float16* __restrict__ emb16) {
  __shared__ float red[256];
  const int row = blockIdx.x, t = threadIdx.x;
  const size_t base = (size_t)row * 256;
  const float x = emb[base + t] + add[base + t];
  red[t] = x; __syncthreads();
  for (int s = 128; s > 0; s >>= 1) { if (t < s) red[t] += red[t + s]; __syncthreads(); }
  const float mean = red[0] * (1.0f / 256.0f); __syncthreads();
  const float d = x - mean;
  red[t] = d * d; __syncthreads();
  for (int s = 128; s > 0; s >>= 1) { if (t < s) red[t] += red[t + s]; __syncthreads(); }
  const float var = red[0] * (1.0f / 256.0f);
  const float y = d * rsqrtf(var + 1e-5f) * g[t] + b[t];
  emb[base + t] = y;
  emb16[base + t] = (_Float16)y;
}

// exact GELU (erf), f32 in -> f16 out
__global__ void __launch_bounds__(256)
k_gelu(const float* __restrict__ in, _Float16* __restrict__ out, int n) {
  const int i = blockIdx.x * 256 + threadIdx.x;
  if (i < n) {
    const float x = in[i];
    out[i] = (_Float16)(0.5f * x * (1.0f + erff(x * 0.70710678118654752f)));
  }
}

// head LN over cls rows: cls16[b,:] = LN(emb[b,0,:]) * g + b
__global__ void __launch_bounds__(256)
k_head_ln(const float* __restrict__ emb, const float* __restrict__ g,
          const float* __restrict__ b, _Float16* __restrict__ cls16) {
  using namespace vit;
  __shared__ float red[256];
  const int bb = blockIdx.x, t = threadIdx.x;
  const float x = emb[((size_t)bb * SEQ) * E + t];
  red[t] = x; __syncthreads();
  for (int s = 128; s > 0; s >>= 1) { if (t < s) red[t] += red[t + s]; __syncthreads(); }
  const float mean = red[0] * (1.0f / 256.0f); __syncthreads();
  const float d = x - mean;
  red[t] = d * d; __syncthreads();
  for (int s = 128; s > 0; s >>= 1) { if (t < s) red[t] += red[t + s]; __syncthreads(); }
  const float var = red[0] * (1.0f / 256.0f);
  cls16[bb * 256 + t] = (_Float16)(d * rsqrtf(var + 1e-5f) * g[t] + b[t]);
}

// ---------------------------------------------------------------------------
extern "C" void kernel_launch(void* const* d_in, const int* in_sizes, int n_in,
                              void* d_out, int out_size, void* d_ws, size_t ws_size,
                              hipStream_t stream) {
  using namespace vit;
  const float* x        = (const float*)d_in[0];
  const float* proj_w   = (const float*)d_in[1];
  const float* proj_b   = (const float*)d_in[2];
  const float* cls_tok  = (const float*)d_in[3];
  const float* pos      = (const float*)d_in[4];
  const float* qkv_w    = (const float*)d_in[5];
  const float* qkv_b    = (const float*)d_in[6];
  const float* out_w    = (const float*)d_in[7];
  const float* out_b    = (const float*)d_in[8];
  const float* ln1_g    = (const float*)d_in[9];
  const float* ln1_b    = (const float*)d_in[10];
  const float* fc1_w    = (const float*)d_in[11];
  const float* fc1_b    = (const float*)d_in[12];
  const float* fc2_w    = (const float*)d_in[13];
  const float* fc2_b    = (const float*)d_in[14];
  const float* ln2_g    = (const float*)d_in[15];
  const float* ln2_b    = (const float*)d_in[16];
  const float* hln_g    = (const float*)d_in[17];
  const float* hln_b    = (const float*)d_in[18];
  const float* head_w   = (const float*)d_in[19];
  const float* head_b   = (const float*)d_in[20];
  float* out            = (float*)d_out;

  // ---- workspace carve (bytes) ----
  char* ws = (char*)d_ws;
  size_t off = 0;
  auto carve = [&](size_t bytes) { char* p = ws + off; off += (bytes + 255) & ~(size_t)255; return p; };
  _Float16* p16    = (_Float16*)carve((size_t)MPATCH * PD * 2);   // 77.1 MB
  float*    pe     = (float*)   carve((size_t)MPATCH * E  * 4);   // 51.4 MB
  float*    emb    = (float*)   carve((size_t)MTOK  * E  * 4);    // 51.6 MB
  _Float16* emb16  = (_Float16*)carve((size_t)MTOK  * E  * 2);    // 25.8 MB
  _Float16* qkv16  = (_Float16*)carve((size_t)MTOK * 3 * E * 2);  // 77.5 MB
  _Float16* o16    = (_Float16*)carve((size_t)MTOK  * E  * 2);    // 25.8 MB
  float*    tmp    = (float*)   carve((size_t)MTOK  * E  * 4);    // 51.6 MB
  _Float16* h16    = (_Float16*)carve((size_t)MTOK  * MLP * 2);   // 25.8 MB
  _Float16* cls16  = (_Float16*)carve((size_t)B * E * 2);
  _Float16* wbuf   = (_Float16*)carve((size_t)3 * E * E * 2);     // max weight (768x256)
  (void)ws_size; (void)n_in; (void)in_sizes; (void)out_size;

  constexpr int NT = 256;
  constexpr size_t ATTN_SMEM = 8192 + 32768 + 32768 + 32768 + 65536;  // 168 KB

  // ---- patch embedding ----
  k_im2col<<<(size_t)MPATCH * PD / NT, NT, 0, stream>>>(x, p16);
  k_cvt<<<(E * PD + NT - 1) / NT, NT, 0, stream>>>(proj_w, wbuf, E * PD);
  k_gemm<false><<<dim3(MPATCH / 128, E / 64), NT, 0, stream>>>(
      p16, PD, wbuf, PD, E, proj_b, (void*)pe, E, PD, E);
  k_embed<<<(size_t)MTOK * E / NT, NT, 0, stream>>>(pe, cls_tok, pos, emb, emb16);

  // ---- transformer layers ----
  for (int i = 0; i < DEPTH; ++i) {
    // QKV projection -> f16 [MTOK, 768]
    k_cvt<<<(3 * E * E + NT - 1) / NT, NT, 0, stream>>>(qkv_w + (size_t)i * 3 * E * E, wbuf, 3 * E * E);
    k_gemm<true><<<dim3(MTOK / 128, (3 * E) / 64), NT, 0, stream>>>(
        emb16, E, wbuf, E, 3 * E, qkv_b + (size_t)i * 3 * E, (void*)qkv16, 3 * E, E, 3 * E);

    // fused attention over the image-batch axis
    k_attn<<<dim3(SEQ * H, B / 64), NT, ATTN_SMEM, stream>>>(qkv16, o16);

    // output projection
    k_cvt<<<(E * E + NT - 1) / NT, NT, 0, stream>>>(out_w + (size_t)i * E * E, wbuf, E * E);
    k_gemm<false><<<dim3(MTOK / 128, E / 64), NT, 0, stream>>>(
        o16, E, wbuf, E, E, out_b + (size_t)i * E, (void*)tmp, E, E, E);
    k_ln_residual<<<MTOK, NT, 0, stream>>>(emb, tmp, ln1_g + (size_t)i * E, ln1_b + (size_t)i * E, emb16);

    // MLP fc1 + GELU
    k_cvt<<<(MLP * E + NT - 1) / NT, NT, 0, stream>>>(fc1_w + (size_t)i * MLP * E, wbuf, MLP * E);
    k_gemm<false><<<dim3(MTOK / 128, MLP / 64), NT, 0, stream>>>(
        emb16, E, wbuf, E, MLP, fc1_b + (size_t)i * MLP, (void*)tmp, MLP, E, MLP);
    k_gelu<<<((size_t)MTOK * MLP + NT - 1) / NT, NT, 0, stream>>>(tmp, h16, MTOK * MLP);

    // MLP fc2
    k_cvt<<<(E * MLP + NT - 1) / NT, NT, 0, stream>>>(fc2_w + (size_t)i * E * MLP, wbuf, E * MLP);
    k_gemm<false><<<dim3(MTOK / 128, E / 64), NT, 0, stream>>>(
        h16, MLP, wbuf, MLP, E, fc2_b + (size_t)i * E, (void*)tmp, E, MLP, E);
    k_ln_residual<<<MTOK, NT, 0, stream>>>(emb, tmp, ln2_g + (size_t)i * E, ln2_b + (size_t)i * E, emb16);
  }

  // ---- classification head ----
  k_head_ln<<<B, NT, 0, stream>>>(emb, hln_g, hln_b, cls16);
  k_cvt<<<(NCLS * E + NT - 1) / NT, NT, 0, stream>>>(head_w, wbuf, NCLS * E);
  k_gemm<false><<<dim3(B / 128, 1), NT, 0, stream>>>(
      cls16, E, wbuf, E, /*wrows=*/NCLS, head_b, (void*)out, NCLS, E, NCLS);
}
